// MambaBlocksSequential_88115549044871
// MI455X (gfx1250) — matrix operand order repaired and checked by reading, not compiled, not run on hardware
//
#include <hip/hip_runtime.h>
#include <cstdint>
#include <cstddef>

// ---------------- constants ----------------
#define NL 4
#define DMODEL 256
#define DI 512
#define DS 16
#define DR 16
#define DC 4
#define BB 4
#define LL 4096
#define MM (BB * LL)   // 16384 rows
#define EPSLN 1e-5f

// ---------------- CDNA5 WMMA types ----------------
typedef __attribute__((ext_vector_type(16))) __bf16 v16bf;
typedef __attribute__((ext_vector_type(8)))  float  v8f;

union Frag32B { uint4 u[2]; v16bf v; };

// ---------------- helpers ----------------
__device__ __forceinline__ float bf2f(uint16_t h) {
    uint32_t u = ((uint32_t)h) << 16;
    float f;
    __builtin_memcpy(&f, &u, 4);
    return f;
}
__device__ __forceinline__ uint16_t f2bf(float f) {
    uint32_t u;
    __builtin_memcpy(&u, &f, 4);
    uint32_t r = (u + 0x7FFFu + ((u >> 16) & 1u)) >> 16;  // RNE
    return (uint16_t)r;
}
__device__ __forceinline__ float wave_reduce_sum(float v) {
    #pragma unroll
    for (int m = 16; m >= 1; m >>= 1) v += __shfl_xor(v, m, 32);
    return v;
}
__device__ __forceinline__ float silu_f(float x) {
    return x / (1.0f + __expf(-x));
}

// ---------------- fused residual add + LayerNorm ----------------
// One wave per row of 256. residual += hidden (optional store), then LN.
// Writes bf16 (per-layer) or f32 (final output).
__global__ void add_ln_kernel(const float* __restrict__ hid,
                              float* __restrict__ res,
                              const float* __restrict__ w,
                              const float* __restrict__ b,
                              uint16_t* __restrict__ out_bf,
                              float* __restrict__ out_f,
                              int writeRes) {
    const int row  = blockIdx.x * 8 + (threadIdx.x >> 5);
    const int lane = threadIdx.x & 31;
    const size_t base = (size_t)row * DMODEL + lane * 8;

    float v[8];
    #pragma unroll
    for (int t = 0; t < 8; t++) v[t] = hid[base + t] + res[base + t];
    if (writeRes) {
        #pragma unroll
        for (int t = 0; t < 8; t++) res[base + t] = v[t];
    }
    float s = 0.f;
    #pragma unroll
    for (int t = 0; t < 8; t++) s += v[t];
    const float mean = wave_reduce_sum(s) * (1.0f / DMODEL);

    float vs = 0.f;
    #pragma unroll
    for (int t = 0; t < 8; t++) { float d = v[t] - mean; vs += d * d; }
    const float var = wave_reduce_sum(vs) * (1.0f / DMODEL);
    const float rstd = rsqrtf(var + EPSLN);

    #pragma unroll
    for (int t = 0; t < 8; t++) {
        const int j = lane * 8 + t;
        const float o = (v[t] - mean) * rstd * w[j] + b[j];
        if (out_bf) out_bf[base + t] = f2bf(o);
        else        out_f[base + t]  = o;
    }
}

// ---------------- f32 -> bf16 convert ----------------
__global__ void f2bf_kernel(const float* __restrict__ in, uint16_t* __restrict__ out, int n) {
    int i = blockIdx.x * blockDim.x + threadIdx.x;
    if (i < n) out[i] = f2bf(in[i]);
}

// ---------------- bf16 WMMA GEMM ----------------
// C[M,N] (f32) = A[M,K](bf16, row-major) * W[N,K](bf16, row-major)^T
// i.e. C[m,n] = sum_k A[m,k] * W[n,k]   (the einsum('bld,ed->ble') pattern).
// One wave owns a 16(M) x (16*NT)(N) strip. NT is compile-time so the
// tile loop fully unrolls: no divergent guards around v_wmma (EXEC stays
// all-ones), and all NT B-fragment loads issue as one clause before the
// NT back-to-back WMMAs (XDL co-execution).
// Requires: K % 32 == 0, N % (16*NT) == 0, M % 16 == 0.
template <int NT>
__global__ void gemm_bf16_wmma(const uint16_t* __restrict__ A,
                               const uint16_t* __restrict__ W,
                               float* __restrict__ C,
                               int M, int N, int K) {
    const int lane = threadIdx.x & 31;
    const int wave = blockIdx.x * (blockDim.x >> 5) + (threadIdx.x >> 5);
    const int nGroups = N / (16 * NT);
    const int mTile  = wave / nGroups;
    const int nGroup = wave - mTile * nGroups;
    if (mTile * 16 >= M) return;

    const int l15 = lane & 15;
    const int hiA = (lane >> 4) << 3;   // 0 or 8  : A prefetch offset / C row offset
    const int hiB = (lane >> 4) << 4;   // 0 or 16 : B K sub-offset

    v8f acc[NT];
    #pragma unroll
    for (int j = 0; j < NT; j++) acc[j] = (v8f){0.f,0.f,0.f,0.f,0.f,0.f,0.f,0.f};

    const uint16_t* __restrict__ Arow = A + (size_t)(mTile * 16 + l15) * K;
    const uint16_t* __restrict__ Wrow[NT];
    #pragma unroll
    for (int j = 0; j < NT; j++)
        Wrow[j] = W + (size_t)((nGroup * NT + j) * 16 + l15) * K + hiB;

    for (int kk = 0; kk < K; kk += 32) {
        // A fragment (16x32 bf16): lane<16 K[kk..+15]; lane>=16 K[kk+16..+31]
        Frag32B a;
        a.u[0] = *(const uint4*)(Arow + kk + hiB);
        a.u[1] = *(const uint4*)(Arow + kk + hiB + 8);
        if (kk + 32 < K)
            __builtin_prefetch(Arow + kk + 32 + hiA, 0, 1);  // global_prefetch_b8

        // All B fragments first (one load clause), then NT WMMAs back-to-back.
        Frag32B bf[NT];
        #pragma unroll
        for (int j = 0; j < NT; j++) {
            bf[j].u[0] = *(const uint4*)(Wrow[j] + kk);       // K = kk+hiB .. +7
            bf[j].u[1] = *(const uint4*)(Wrow[j] + kk + 8);   // K = kk+hiB+8 .. +15
        }
        #pragma unroll
        for (int j = 0; j < NT; j++)
            acc[j] = __builtin_amdgcn_wmma_f32_16x16x32_bf16(
                false, a.v, false, bf[j].v, (short)0, acc[j], false, false);
    }

    const int rowBase = mTile * 16 + hiA;  // lanes>=16 hold M = 8+r
    #pragma unroll
    for (int j = 0; j < NT; j++) {
        const int n = (nGroup * NT + j) * 16 + l15;
        #pragma unroll
        for (int r = 0; r < 8; r++)
            C[(size_t)(rowBase + r) * N + n] = acc[j][r];
    }
}

// ---------------- depthwise causal conv (width 4) + SiLU ----------------
// reads xs = xz[:, :DI]; writes u in bf16
__global__ void conv_silu_kernel(const float* __restrict__ xz,
                                 const float* __restrict__ cw,
                                 const float* __restrict__ cb,
                                 uint16_t* __restrict__ u_bf) {
    const int idx = blockIdx.x * blockDim.x + threadIdx.x;  // over MM*DI
    const int c = idx & (DI - 1);
    const int m = idx >> 9;
    const int l = m & (LL - 1);
    const int b = m >> 12;
    float acc = cb[c];
    #pragma unroll
    for (int j = 0; j < DC; j++) {
        const int l2 = l - (DC - 1) + j;
        if (l2 >= 0)
            acc += xz[(size_t)((b << 12) + l2) * (2 * DI) + c] * cw[c * DC + j];
    }
    u_bf[idx] = f2bf(silu_f(acc));
}

// ---------------- delta = softplus(xdbl[:, :DR] @ Wdt^T + bdt) ----------------
__global__ void dt_kernel(const float* __restrict__ xdbl,
                          const float* __restrict__ Wdt,
                          const float* __restrict__ bdt,
                          float* __restrict__ delta) {
    const int idx = blockIdx.x * blockDim.x + threadIdx.x;  // over MM*DI
    const int d = idx & (DI - 1);
    const int m = idx >> 9;
    const float* __restrict__ xr = xdbl + (size_t)m * (DR + 2 * DS);
    float acc = bdt[d];
    #pragma unroll
    for (int r = 0; r < DR; r++) acc += xr[r] * Wdt[d * DR + r];
    delta[idx] = (acc > 20.f) ? acc : log1pf(__expf(acc));
}

// ---------------- selective scan: one lane per (b, d) channel ----------------
__global__ void scan_kernel(const float* __restrict__ delta,
                            const uint16_t* __restrict__ u_bf,
                            const float* __restrict__ xdbl,
                            const float* __restrict__ Alog,
                            const float* __restrict__ Dp,
                            float* __restrict__ y) {
    const int tid = blockIdx.x * blockDim.x + threadIdx.x;  // 2048 = BB*DI
    const int d = tid & (DI - 1);
    const int b = tid >> 9;

    float A[DS], h[DS];
    #pragma unroll
    for (int s = 0; s < DS; s++) {
        A[s] = -__expf(Alog[d * DS + s]);
        h[s] = 0.f;
    }
    const float Dd = Dp[d];
    const size_t mb = (size_t)b * LL;

    for (int t = 0; t < LL; t++) {
        const size_t m = mb + t;
        const float dt = delta[m * DI + d];
        const float ut = bf2f(u_bf[m * DI + d]);
        const float du = dt * ut;
        const float* __restrict__ xd = xdbl + m * (DR + 2 * DS);
        float acc = 0.f;
        #pragma unroll
        for (int s = 0; s < DS; s++) {
            const float e = __expf(dt * A[s]);
            h[s] = e * h[s] + du * xd[DR + s];          // B_t[s]
            acc += h[s] * xd[DR + DS + s];              // C_t[s]
        }
        y[m * DI + d] = acc + ut * Dd;
    }
}

// ---------------- gate: y *= silu(z); -> bf16 ----------------
__global__ void gate_kernel(const float* __restrict__ y,
                            const float* __restrict__ xz,
                            uint16_t* __restrict__ y_bf) {
    const int idx = blockIdx.x * blockDim.x + threadIdx.x;  // over MM*DI
    const int d = idx & (DI - 1);
    const int m = idx >> 9;
    const float z = xz[(size_t)m * (2 * DI) + DI + d];
    y_bf[idx] = f2bf(y[idx] * silu_f(z));
}

// ---------------- host launch ----------------
extern "C" void kernel_launch(void* const* d_in, const int* in_sizes, int n_in,
                              void* d_out, int out_size, void* d_ws, size_t ws_size,
                              hipStream_t stream) {
    (void)in_sizes; (void)n_in; (void)out_size; (void)ws_size;

    const float* x         = (const float*)d_in[0];
    const float* norm_w    = (const float*)d_in[1];
    const float* norm_b    = (const float*)d_in[2];
    const float* in_proj_w = (const float*)d_in[3];
    const float* conv_w    = (const float*)d_in[4];
    const float* conv_b    = (const float*)d_in[5];
    const float* x_proj_w  = (const float*)d_in[6];
    const float* dt_proj_w = (const float*)d_in[7];
    const float* dt_proj_b = (const float*)d_in[8];
    const float* A_log     = (const float*)d_in[9];
    const float* D_ssm     = (const float*)d_in[10];
    const float* out_proj_w= (const float*)d_in[11];
    const float* norm_f_w  = (const float*)d_in[12];
    const float* norm_f_b  = (const float*)d_in[13];

    char* base = (char*)d_ws;
    auto alloc = [&](size_t bytes) -> void* {
        void* p = (void*)base;
        base += (bytes + 255) & ~(size_t)255;
        return p;
    };

    float*    residual = (float*)   alloc((size_t)MM * DMODEL * 4);
    float*    hidden   = (float*)   alloc((size_t)MM * DMODEL * 4);
    float*    xz       = (float*)   alloc((size_t)MM * 2 * DI * 4);
    float*    delta    = (float*)   alloc((size_t)MM * DI * 4);
    float*    ybuf     = (float*)   alloc((size_t)MM * DI * 4);
    float*    xdbl     = (float*)   alloc((size_t)MM * (DR + 2 * DS) * 4);
    uint16_t* h_bf     = (uint16_t*)alloc((size_t)MM * DMODEL * 2);
    uint16_t* u_bf     = (uint16_t*)alloc((size_t)MM * DI * 2);
    uint16_t* y_bf     = (uint16_t*)alloc((size_t)MM * DI * 2);
    uint16_t* Wi_bf    = (uint16_t*)alloc((size_t)(2 * DI) * DMODEL * 2);
    uint16_t* Wx_bf    = (uint16_t*)alloc((size_t)(DR + 2 * DS) * DI * 2);
    uint16_t* Wo_bf    = (uint16_t*)alloc((size_t)DMODEL * DI * 2);

    hipMemcpyAsync(hidden, x, (size_t)MM * DMODEL * 4, hipMemcpyDeviceToDevice, stream);
    hipMemsetAsync(residual, 0, (size_t)MM * DMODEL * 4, stream);

    const int ELT_BLOCKS = (MM * DI) / 256;  // elementwise grids over MM*DI

    for (int i = 0; i < NL; i++) {
        // residual += hidden; h = LN(residual) -> bf16
        add_ln_kernel<<<MM / 8, 256, 0, stream>>>(
            hidden, residual, norm_w + i * DMODEL, norm_b + i * DMODEL,
            h_bf, nullptr, 1);

        // convert this layer's GEMM weights to bf16
        f2bf_kernel<<<(2 * DI * DMODEL + 255) / 256, 256, 0, stream>>>(
            in_proj_w + (size_t)i * 2 * DI * DMODEL, Wi_bf, 2 * DI * DMODEL);
        f2bf_kernel<<<((DR + 2 * DS) * DI + 255) / 256, 256, 0, stream>>>(
            x_proj_w + (size_t)i * (DR + 2 * DS) * DI, Wx_bf, (DR + 2 * DS) * DI);
        f2bf_kernel<<<(DMODEL * DI + 255) / 256, 256, 0, stream>>>(
            out_proj_w + (size_t)i * DMODEL * DI, Wo_bf, DMODEL * DI);

        // xz = h @ Wi^T : M=16384, N=1024 (NT=4 -> 16 groups), K=256
        {
            int waves = (MM / 16) * (2 * DI / 64);
            gemm_bf16_wmma<4><<<(waves * 32) / 256, 256, 0, stream>>>(
                h_bf, Wi_bf, xz, MM, 2 * DI, DMODEL);
        }

        // u = silu(causal_conv(xs)) -> bf16
        conv_silu_kernel<<<ELT_BLOCKS, 256, 0, stream>>>(
            xz, conv_w + (size_t)i * DI * DC, conv_b + (size_t)i * DI, u_bf);

        // xdbl = u @ Wx^T : M=16384, N=48 (NT=3 -> 1 group), K=512
        {
            int waves = (MM / 16) * 1;
            gemm_bf16_wmma<3><<<(waves * 32) / 256, 256, 0, stream>>>(
                u_bf, Wx_bf, xdbl, MM, DR + 2 * DS, DI);
        }

        // delta = softplus(xdbl[:, :DR] @ Wdt^T + bdt)
        dt_kernel<<<ELT_BLOCKS, 256, 0, stream>>>(
            xdbl, dt_proj_w + (size_t)i * DI * DR, dt_proj_b + (size_t)i * DI, delta);

        // selective scan -> y (includes + u * D)
        scan_kernel<<<(BB * DI) / 256, 256, 0, stream>>>(
            delta, u_bf, xdbl, A_log + (size_t)i * DI * DS, D_ssm + (size_t)i * DI, ybuf);

        // y *= silu(z) -> bf16
        gate_kernel<<<ELT_BLOCKS, 256, 0, stream>>>(ybuf, xz, y_bf);

        // hidden = y @ Wo^T : M=16384, N=256 (NT=4 -> 4 groups), K=512
        {
            int waves = (MM / 16) * (DMODEL / 64);
            gemm_bf16_wmma<4><<<(waves * 32) / 256, 256, 0, stream>>>(
                y_bf, Wo_bf, hidden, MM, DMODEL, DI);
        }
    }

    // final: residual += hidden; out = LN(residual, norm_f) -> f32 d_out
    add_ln_kernel<<<MM / 8, 256, 0, stream>>>(
        hidden, residual, norm_f_w, norm_f_b, nullptr, (float*)d_out, 0);
}
